// MultiHeadAttention_59459527246413
// MI455X (gfx1250) — matrix operand order repaired and checked by reading, not compiled, not run on hardware
//
#include <hip/hip_runtime.h>
#include <hip/hip_bf16.h>
#include <stdint.h>

#define EMBED 1024
#define HEADS 16
#define HEAD_DIM 64
#define NB 4
#define SEQ 2048
#define MROWS (NB * SEQ) /* 8192 */

typedef __bf16 bf16_t;
typedef __bf16 v16bf __attribute__((ext_vector_type(16)));
typedef __bf16 v8bf  __attribute__((ext_vector_type(8)));
typedef float  v8f   __attribute__((ext_vector_type(8)));

static __device__ __forceinline__ v8f wmma_bf16(v16bf a, v16bf b, v8f c) {
  // D = A(16x32 bf16) * B(32x16 bf16) + C(16x16 f32)
  return __builtin_amdgcn_wmma_f32_16x16x32_bf16(false, a, false, b, (short)0, c,
                                                 false, false);
}

// A-fragment (16x32, bf16): lane holds row M = lane&15; K chunk base (lane>>4)*16,
// elems 0..7 at +0, elems 8..15 at +8 (16 contiguous K, matching load_fragB).
static __device__ __forceinline__ v16bf load_fragA(const bf16_t* base, int row0,
                                                   int ld, int k0, int lane) {
  const bf16_t* p = base + (size_t)(row0 + (lane & 15)) * ld + k0 + ((lane >> 4) << 4);
  v8bf lo = *(const v8bf*)p;
  v8bf hi = *(const v8bf*)(p + 8);
  v16bf r;
#pragma unroll
  for (int i = 0; i < 8; ++i) { r[i] = lo[i]; r[i + 8] = hi[i]; }
  return r;
}

// B-fragment (32x16, bf16): lane holds col N = lane&15; 16 contiguous K values
// starting at k0 + (lane>>4)*16. Source is row-major [Ncols, K] (i.e. B = W^T).
static __device__ __forceinline__ v16bf load_fragB(const bf16_t* base, int col0,
                                                   int ld, int k0, int lane) {
  const bf16_t* p = base + (size_t)(col0 + (lane & 15)) * ld + k0 + ((lane >> 4) << 4);
  return *(const v16bf*)p;
}

// f32 -> bf16 conversion with optional scale (folds 1/sqrt(D) into Wq).
__global__ void cvt_bf16_kernel(const float* __restrict__ src, bf16_t* __restrict__ dst,
                                float scale, long nElem) {
  long i = (long)blockIdx.x * blockDim.x + threadIdx.x;
  long stride = (long)gridDim.x * blockDim.x;
  for (; i < nElem; i += stride) dst[i] = (bf16_t)(src[i] * scale);
}

// C = A[M,K] @ W[Ncols,K]^T + biasScale*bias.  4 waves/block; each wave owns a
// 32x64 tile via 2x4 WMMA register blocking (8 WMMAs per K-step, A frags reused
// 4x, B frags reused 2x).  storeMode: 0 -> bf16 [N,H,L,D], 1 -> bf16 [N,H,D,L]
// (V transposed), 2 -> f32 row-major (final output).
__global__ __launch_bounds__(128) void gemm_xwT_kernel(
    const bf16_t* __restrict__ A, const bf16_t* __restrict__ W,
    const float* __restrict__ bias, float biasScale, void* __restrict__ outp,
    int K, int Ncols, int storeMode) {
  const int lane = threadIdx.x & 31;
  const int wave = threadIdx.x >> 5;
  const int row0 = blockIdx.y * 32;
  const int col0 = blockIdx.x * 256 + wave * 64;

  v8f zero = {0, 0, 0, 0, 0, 0, 0, 0};
  v8f acc[2][4];
#pragma unroll
  for (int i = 0; i < 2; ++i)
#pragma unroll
    for (int j = 0; j < 4; ++j) acc[i][j] = zero;

  for (int k0 = 0; k0 < K; k0 += 32) {
    v16bf a0 = load_fragA(A, row0, K, k0, lane);
    v16bf a1 = load_fragA(A, row0 + 16, K, k0, lane);
    v16bf b[4];
#pragma unroll
    for (int j = 0; j < 4; ++j) b[j] = load_fragB(W, col0 + j * 16, K, k0, lane);
    if (k0 + 128 < K)  // read-ahead A into near caches (WGP scope)
      __builtin_prefetch((const void*)(A + (size_t)(row0 + (lane & 15)) * K + k0 + 128), 0, 3);
#pragma unroll
    for (int j = 0; j < 4; ++j) {
      acc[0][j] = wmma_bf16(a0, b[j], acc[0][j]);
      acc[1][j] = wmma_bf16(a1, b[j], acc[1][j]);
    }
  }

  const int nIdx = lane & 15;
  const int mBase = (lane >> 4) << 3;
#pragma unroll
  for (int ti = 0; ti < 2; ++ti) {
#pragma unroll
    for (int tj = 0; tj < 4; ++tj) {
#pragma unroll
      for (int v = 0; v < 8; ++v) {
        int r = row0 + ti * 16 + mBase + v;
        int c = col0 + tj * 16 + nIdx;
        float val = acc[ti][tj][v] + biasScale * bias[c];
        if (storeMode == 2) {
          ((float*)outp)[(size_t)r * Ncols + c] = val;
        } else {
          int n = r >> 11, l = r & (SEQ - 1);
          int h = c >> 6, d = c & (HEAD_DIM - 1);
          bf16_t* o = (bf16_t*)outp;
          if (storeMode == 0)
            o[(((size_t)(n * HEADS + h)) * SEQ + l) * HEAD_DIM + d] = (bf16_t)val;
          else
            o[(((size_t)(n * HEADS + h)) * HEAD_DIM + d) * SEQ + l] = (bf16_t)val;
        }
      }
    }
  }
}

// Pass 1: streaming online softmax stats (row max m, row sum-exp z) per (n,h,q).
__global__ __launch_bounds__(32) void attn_stats_kernel(
    const bf16_t* __restrict__ Q, const bf16_t* __restrict__ Kh,
    const float* __restrict__ bias, const unsigned char* __restrict__ mask,
    float* __restrict__ mOut, float* __restrict__ zOut) {
  const int lane = threadIdx.x & 31;
  const int q0 = blockIdx.x * 16;
  const int h = blockIdx.y;
  const int n = blockIdx.z;
  const bf16_t* Qh = Q + ((size_t)(n * HEADS + h)) * SEQ * HEAD_DIM;
  const bf16_t* Kp = Kh + ((size_t)(n * HEADS + h)) * SEQ * HEAD_DIM;
  const int nIdx = lane & 15;
  const int mBase = (lane >> 4) << 3;

  v16bf aq0 = load_fragA(Qh, q0, HEAD_DIM, 0, lane);
  v16bf aq1 = load_fragA(Qh, q0, HEAD_DIM, 32, lane);

  float mrun[8], zrun[8];
#pragma unroll
  for (int v = 0; v < 8; ++v) { mrun[v] = -3.0e38f; zrun[v] = 0.0f; }

  for (int k0 = 0; k0 < SEQ; k0 += 16) {
    v16bf b0 = load_fragB(Kp, k0, HEAD_DIM, 0, lane);
    v16bf b1 = load_fragB(Kp, k0, HEAD_DIM, 32, lane);
    v8f s = {0, 0, 0, 0, 0, 0, 0, 0};
    s = wmma_bf16(aq0, b0, s);
    s = wmma_bf16(aq1, b1, s);
    const int kc = k0 + nIdx;
    const bool masked = mask[(size_t)n * SEQ + kc] != 0;
    const size_t bbase = ((size_t)h * SEQ + (q0 + mBase)) * SEQ + kc;
#pragma unroll
    for (int v = 0; v < 8; ++v) {
      float sv = s[v] + bias[bbase + (size_t)v * SEQ];
      if (masked) sv = -1.0e30f;
      float tmax = sv;  // row max over the 16 lanes holding this row
      tmax = fmaxf(tmax, __shfl_xor(tmax, 1, 32));
      tmax = fmaxf(tmax, __shfl_xor(tmax, 2, 32));
      tmax = fmaxf(tmax, __shfl_xor(tmax, 4, 32));
      tmax = fmaxf(tmax, __shfl_xor(tmax, 8, 32));
      float mnew = fmaxf(mrun[v], tmax);
      float e = __expf(sv - mnew);
      e += __shfl_xor(e, 1, 32);
      e += __shfl_xor(e, 2, 32);
      e += __shfl_xor(e, 4, 32);
      e += __shfl_xor(e, 8, 32);
      zrun[v] = zrun[v] * __expf(mrun[v] - mnew) + e;
      mrun[v] = mnew;
    }
  }
  if (nIdx == 0) {
    size_t base = ((size_t)(n * HEADS + h)) * SEQ + q0 + mBase;
#pragma unroll
    for (int v = 0; v < 8; ++v) { mOut[base + v] = mrun[v]; zOut[base + v] = zrun[v]; }
  }
}

// Pass 2: head-averaged attention weights, one deterministic write per tile
// (loop over heads in-register; no float atomics).
__global__ __launch_bounds__(32) void attn_avg_kernel(
    const bf16_t* __restrict__ Q, const bf16_t* __restrict__ Kh,
    const float* __restrict__ bias, const unsigned char* __restrict__ mask,
    const float* __restrict__ mSt, const float* __restrict__ zSt,
    float* __restrict__ avgOut) {
  const int lane = threadIdx.x & 31;
  const int k0 = blockIdx.x * 16;
  const int q0 = blockIdx.y * 16;
  const int n = blockIdx.z;
  const int nIdx = lane & 15;
  const int mBase = (lane >> 4) << 3;
  const bool masked = mask[(size_t)n * SEQ + k0 + nIdx] != 0;

  float acc[8];
#pragma unroll
  for (int v = 0; v < 8; ++v) acc[v] = 0.0f;

  for (int h = 0; h < HEADS; ++h) {
    const bf16_t* Qh = Q + ((size_t)(n * HEADS + h)) * SEQ * HEAD_DIM;
    const bf16_t* Kp = Kh + ((size_t)(n * HEADS + h)) * SEQ * HEAD_DIM;
    v16bf aq0 = load_fragA(Qh, q0, HEAD_DIM, 0, lane);
    v16bf aq1 = load_fragA(Qh, q0, HEAD_DIM, 32, lane);
    v16bf b0 = load_fragB(Kp, k0, HEAD_DIM, 0, lane);
    v16bf b1 = load_fragB(Kp, k0, HEAD_DIM, 32, lane);
    v8f s = {0, 0, 0, 0, 0, 0, 0, 0};
    s = wmma_bf16(aq0, b0, s);
    s = wmma_bf16(aq1, b1, s);
    const size_t bbase = ((size_t)h * SEQ + (q0 + mBase)) * SEQ + k0 + nIdx;
    const size_t sbase = ((size_t)(n * HEADS + h)) * SEQ + q0 + mBase;
#pragma unroll
    for (int v = 0; v < 8; ++v) {
      float mm = mSt[sbase + v];
      float zz = zSt[sbase + v];
      float inv = (zz > 0.0f) ? (1.0f / zz) : 0.0f;
      float sv = s[v] + bias[bbase + (size_t)v * SEQ];
      float p = masked ? 0.0f : __expf(sv - mm) * inv;
      acc[v] += p;
    }
  }
#pragma unroll
  for (int v = 0; v < 8; ++v)
    avgOut[((size_t)n * SEQ + q0 + mBase + v) * SEQ + k0 + nIdx] =
        acc[v] * (1.0f / HEADS);
}

// Pass 3: out = P @ V. Recompute score tiles, normalize with stats, stage P
// through LDS (C-layout -> A-fragment layout), WMMA against V^T [N,H,D,L].
__global__ __launch_bounds__(32) void attn_out_kernel(
    const bf16_t* __restrict__ Q, const bf16_t* __restrict__ Kh,
    const bf16_t* __restrict__ Vt, const float* __restrict__ bias,
    const unsigned char* __restrict__ mask, const float* __restrict__ mSt,
    const float* __restrict__ zSt, bf16_t* __restrict__ attnCat) {
  __shared__ bf16_t pTile[16 * 32];
  const int lane = threadIdx.x & 31;
  const int q0 = blockIdx.x * 16;
  const int h = blockIdx.y;
  const int n = blockIdx.z;
  const int nIdx = lane & 15;
  const int mBase = (lane >> 4) << 3;
  const bf16_t* Qh = Q + ((size_t)(n * HEADS + h)) * SEQ * HEAD_DIM;
  const bf16_t* Kp = Kh + ((size_t)(n * HEADS + h)) * SEQ * HEAD_DIM;
  const bf16_t* Vh = Vt + ((size_t)(n * HEADS + h)) * HEAD_DIM * SEQ;

  v16bf aq0 = load_fragA(Qh, q0, HEAD_DIM, 0, lane);
  v16bf aq1 = load_fragA(Qh, q0, HEAD_DIM, 32, lane);

  float mrow[8], invz[8];
  {
    size_t sbase = ((size_t)(n * HEADS + h)) * SEQ + q0 + mBase;
#pragma unroll
    for (int v = 0; v < 8; ++v) {
      mrow[v] = mSt[sbase + v];
      float zz = zSt[sbase + v];
      invz[v] = (zz > 0.0f) ? (1.0f / zz) : 0.0f;
    }
  }

  v8f zero = {0, 0, 0, 0, 0, 0, 0, 0};
  v8f acc[4] = {zero, zero, zero, zero};

  for (int k0 = 0; k0 < SEQ; k0 += 32) {
#pragma unroll
    for (int half = 0; half < 2; ++half) {
      const int kk = k0 + half * 16;
      v16bf b0 = load_fragB(Kp, kk, HEAD_DIM, 0, lane);
      v16bf b1 = load_fragB(Kp, kk, HEAD_DIM, 32, lane);
      v8f s = zero;
      s = wmma_bf16(aq0, b0, s);
      s = wmma_bf16(aq1, b1, s);
      const bool masked = mask[(size_t)n * SEQ + kk + nIdx] != 0;
      const size_t bbase = ((size_t)h * SEQ + (q0 + mBase)) * SEQ + kk + nIdx;
#pragma unroll
      for (int v = 0; v < 8; ++v) {
        float sv = s[v] + bias[bbase + (size_t)v * SEQ];
        float p = masked ? 0.0f : __expf(sv - mrow[v]) * invz[v];
        pTile[(mBase + v) * 32 + half * 16 + nIdx] = (bf16_t)p;
      }
    }
    __syncthreads();
    v16bf pa = load_fragA(pTile, 0, 32, 0, lane);  // P as 16x32 A-fragment (ds loads)
#pragma unroll
    for (int dt = 0; dt < 4; ++dt) {
      v16bf bv = load_fragB(Vh, dt * 16, SEQ, k0, lane);  // B[k,d] = V^T[d,k]
      acc[dt] = wmma_bf16(pa, bv, acc[dt]);
    }
    __syncthreads();
  }

#pragma unroll
  for (int dt = 0; dt < 4; ++dt)
#pragma unroll
    for (int v = 0; v < 8; ++v)
      attnCat[((size_t)n * SEQ + q0 + mBase + v) * EMBED + h * HEAD_DIM +
              dt * 16 + nIdx] = (bf16_t)acc[dt][v];
}

extern "C" void kernel_launch(void* const* d_in, const int* in_sizes, int n_in,
                              void* d_out, int out_size, void* d_ws, size_t ws_size,
                              hipStream_t stream) {
  (void)in_sizes; (void)n_in; (void)out_size; (void)ws_size;
  const float* query = (const float*)d_in[0];
  const float* attn_bias = (const float*)d_in[1];
  const unsigned char* mask = (const unsigned char*)d_in[2];
  const float* Wq = (const float*)d_in[3];
  const float* bq = (const float*)d_in[4];
  const float* Wk = (const float*)d_in[5];
  const float* bk = (const float*)d_in[6];
  const float* Wv = (const float*)d_in[7];
  const float* bv = (const float*)d_in[8];
  const float* Wo = (const float*)d_in[9];
  const float* bo = (const float*)d_in[10];

  // Workspace carve-up (~93 MB total)
  char* w = (char*)d_ws;
  size_t off = 0;
  auto carve = [&](size_t bytes) {
    void* p = (void*)(w + off);
    off = (off + bytes + 255) & ~(size_t)255;
    return p;
  };
  bf16_t* xb  = (bf16_t*)carve((size_t)MROWS * EMBED * 2);   // query, bf16
  bf16_t* wqB = (bf16_t*)carve((size_t)EMBED * EMBED * 2);   // pre-scaled by 1/sqrt(D)
  bf16_t* wkB = (bf16_t*)carve((size_t)EMBED * EMBED * 2);
  bf16_t* wvB = (bf16_t*)carve((size_t)EMBED * EMBED * 2);
  bf16_t* woB = (bf16_t*)carve((size_t)EMBED * EMBED * 2);
  bf16_t* qh  = (bf16_t*)carve((size_t)NB * HEADS * SEQ * HEAD_DIM * 2);  // [N,H,L,D]
  bf16_t* kh  = (bf16_t*)carve((size_t)NB * HEADS * SEQ * HEAD_DIM * 2);  // [N,H,L,D]
  bf16_t* vt  = (bf16_t*)carve((size_t)NB * HEADS * HEAD_DIM * SEQ * 2);  // [N,H,D,L]
  bf16_t* cat = (bf16_t*)carve((size_t)MROWS * EMBED * 2);   // attn concat [N,L,E]
  float*  mSt = (float*)carve((size_t)NB * HEADS * SEQ * 4);
  float*  zSt = (float*)carve((size_t)NB * HEADS * SEQ * 4);

  const float scaling = 0.125f;  // HEAD_DIM^-0.5

  cvt_bf16_kernel<<<2048, 256, 0, stream>>>(query, xb, 1.0f, (long)MROWS * EMBED);
  cvt_bf16_kernel<<<512, 256, 0, stream>>>(Wq, wqB, scaling, (long)EMBED * EMBED);
  cvt_bf16_kernel<<<512, 256, 0, stream>>>(Wk, wkB, 1.0f, (long)EMBED * EMBED);
  cvt_bf16_kernel<<<512, 256, 0, stream>>>(Wv, wvB, 1.0f, (long)EMBED * EMBED);
  cvt_bf16_kernel<<<512, 256, 0, stream>>>(Wo, woB, 1.0f, (long)EMBED * EMBED);

  dim3 gGemm(EMBED / 256, MROWS / 32);  // (4, 256), 128 threads = 4 waves (32x64/wave)
  gemm_xwT_kernel<<<gGemm, 128, 0, stream>>>(xb, wqB, bq, scaling, qh, EMBED, EMBED, 0);
  gemm_xwT_kernel<<<gGemm, 128, 0, stream>>>(xb, wkB, bk, 1.0f, kh, EMBED, EMBED, 0);
  gemm_xwT_kernel<<<gGemm, 128, 0, stream>>>(xb, wvB, bv, 1.0f, vt, EMBED, EMBED, 1);

  dim3 gStats(SEQ / 16, HEADS, NB);
  attn_stats_kernel<<<gStats, 32, 0, stream>>>(qh, kh, attn_bias, mask, mSt, zSt);

  float* avgOut = (float*)d_out + (size_t)NB * SEQ * EMBED;
  dim3 gAvg(SEQ / 16, SEQ / 16, NB);
  attn_avg_kernel<<<gAvg, 32, 0, stream>>>(qh, kh, attn_bias, mask, mSt, zSt, avgOut);

  dim3 gOut(SEQ / 16, HEADS, NB);
  attn_out_kernel<<<gOut, 32, 0, stream>>>(qh, kh, vt, attn_bias, mask, mSt, zSt, cat);

  gemm_xwT_kernel<<<gGemm, 128, 0, stream>>>(cat, woB, bo, 1.0f, d_out, EMBED, EMBED, 2);
}
